// QuantumAttention_1743756722689
// MI455X (gfx1250) — matrix operand order repaired and checked by reading, not compiled, not run on hardware
//
#include <hip/hip_runtime.h>
#include <hip/hip_bf16.h>

typedef __attribute__((ext_vector_type(16))) __bf16 v16bf;
typedef __attribute__((ext_vector_type(8)))  float  v8f;

union V16 { v16bf v; uint4 u[2]; };

#define T_DIM 2048
#define S_DIM 2048
#define B_DIM 8
#define C_DIM 256
#define M_TILE 32

// ---------------------------------------------------------------------------
// Row-normalize (T,B,C) f32 -> (B,T,C) bf16 unit rows.  One wave per row.
// ---------------------------------------------------------------------------
__global__ __launch_bounds__(256) void qa_norm_kernel(const float* __restrict__ x,
                                                      __bf16* __restrict__ out,
                                                      int outerOut) {
  const int wave = threadIdx.x >> 5;
  const int lane = threadIdx.x & 31;
  const int row  = blockIdx.x * 8 + wave;     // row index in (t*B + b)
  const int t = row / B_DIM;
  const int b = row % B_DIM;

  const float* xr = x + (size_t)row * C_DIM + lane * 8;
  float4 f0 = *reinterpret_cast<const float4*>(xr);
  float4 f1 = *reinterpret_cast<const float4*>(xr + 4);

  float ss = f0.x*f0.x + f0.y*f0.y + f0.z*f0.z + f0.w*f0.w
           + f1.x*f1.x + f1.y*f1.y + f1.z*f1.z + f1.w*f1.w;
  ss += __shfl_xor(ss, 1, 32);
  ss += __shfl_xor(ss, 2, 32);
  ss += __shfl_xor(ss, 4, 32);
  ss += __shfl_xor(ss, 8, 32);
  ss += __shfl_xor(ss, 16, 32);

  const float scale = 1.0f / fmaxf(sqrtf(ss), 1e-8f);

  __attribute__((aligned(16))) __bf16 o[8];
  o[0] = (__bf16)(f0.x * scale); o[1] = (__bf16)(f0.y * scale);
  o[2] = (__bf16)(f0.z * scale); o[3] = (__bf16)(f0.w * scale);
  o[4] = (__bf16)(f1.x * scale); o[5] = (__bf16)(f1.y * scale);
  o[6] = (__bf16)(f1.z * scale); o[7] = (__bf16)(f1.w * scale);

  *reinterpret_cast<uint4*>(out + ((size_t)b * outerOut + t) * C_DIM + lane * 8)
      = *reinterpret_cast<uint4*>(o);
}

// ---------------------------------------------------------------------------
// Transpose value (S,B,C) f32 -> vT (B,C,S) bf16 via 64x64 LDS tiles, so the
// PV WMMA B-operand is contiguous-per-lane in the documented B layout.
// ---------------------------------------------------------------------------
__global__ __launch_bounds__(256) void qa_vt_kernel(const float* __restrict__ val,
                                                    __bf16* __restrict__ vT) {
  __shared__ __bf16 tile[64][72];             // padded to dodge bank conflicts
  const int nc  = C_DIM / 64;                 // 4
  const int ns  = S_DIM / 64;                 // 32
  const int b   = blockIdx.x / (ns * nc);
  const int rem = blockIdx.x % (ns * nc);
  const int ts  = (rem / nc) * 64;
  const int tc  = (rem % nc) * 64;

  const int tid = threadIdx.x;
  const int r = tid >> 2;                     // 0..63
  const int q = tid & 3;                      // 16-col segment

  const float* src = val + ((size_t)(ts + r) * B_DIM + b) * C_DIM + tc + q * 16;
  #pragma unroll
  for (int i = 0; i < 4; ++i) {
    float4 f = reinterpret_cast<const float4*>(src)[i];
    tile[q*16 + i*4 + 0][r] = (__bf16)f.x;
    tile[q*16 + i*4 + 1][r] = (__bf16)f.y;
    tile[q*16 + i*4 + 2][r] = (__bf16)f.z;
    tile[q*16 + i*4 + 3][r] = (__bf16)f.w;
  }
  __syncthreads();

  __attribute__((aligned(16))) __bf16 o[16];
  #pragma unroll
  for (int i = 0; i < 16; ++i) o[i] = tile[r][q*16 + i];
  uint4* dst = reinterpret_cast<uint4*>(vT + ((size_t)b * C_DIM + tc + r) * S_DIM + ts + q * 16);
  dst[0] = reinterpret_cast<uint4*>(o)[0];
  dst[1] = reinterpret_cast<uint4*>(o)[1];
}

// ---------------------------------------------------------------------------
// Fused attention: one workgroup = (batch b, 32 query rows).
//   scores -> square -> exp -> mask   (bf16 WMMA, single-pass softmax:
//   scores in [0,1] so no running max needed; all-masked row => sum==0)
//   unnormalized probs kept in LDS (32 x 2048 bf16 = 128KB of the 320KB WGP LDS)
//   then (a) attn weights streamed out f32, (b) PV WMMA from LDS probs.
// ---------------------------------------------------------------------------
__global__ __launch_bounds__(256) void qa_attn_kernel(
    const __bf16* __restrict__ qhat, const __bf16* __restrict__ khat,
    const __bf16* __restrict__ vT,   const unsigned char* __restrict__ mask,
    float* __restrict__ ctx, float* __restrict__ attn) {
  extern __shared__ char smem[];
  __bf16* q_lds  = reinterpret_cast<__bf16*>(smem);                         // 32x256
  __bf16* p_lds  = reinterpret_cast<__bf16*>(smem + M_TILE*C_DIM*2);        // 32x2048
  float*  rowsum = reinterpret_cast<float*>(smem + M_TILE*C_DIM*2 + M_TILE*S_DIM*2);
  float*  invz   = rowsum + M_TILE;

  const int tid    = threadIdx.x;
  const int wave   = tid >> 5;
  const int lane   = tid & 31;
  const int l15    = lane & 15;
  const int lhalf  = lane >> 4;
  const int base_k = lhalf ? 8 : 0;           // A-layout K offset per lane half

  const int b  = blockIdx.x / (T_DIM / M_TILE);
  const int m0 = (blockIdx.x % (T_DIM / M_TILE)) * M_TILE;

  // ---- stage normalized Q tile into LDS (32 rows x 256 bf16) ----
  {
    const int r = tid >> 3, seg = tid & 7;
    const uint4* src = reinterpret_cast<const uint4*>(
        qhat + ((size_t)b * T_DIM + m0 + r) * C_DIM + seg * 32);
    uint4* dst = reinterpret_cast<uint4*>(q_lds + r * C_DIM + seg * 32);
    dst[0] = src[0]; dst[1] = src[1]; dst[2] = src[2]; dst[3] = src[3];
  }
  if (tid < M_TILE) rowsum[tid] = 0.0f;
  __syncthreads();

  const int mt = (wave & 1) * 16;             // which 16-row half of the 32-row tile
  const int ng = wave >> 1;                   // column-group (0..3)

  // ---- preload Q A-fragments for all 8 K-chunks (reused for all 128 S-tiles)
  // A 16x32 bf16 layout: lane m=l15, elems e: K = base_k + e + (e>=8 ? 8 : 0)
  v16bf aq[8];
  #pragma unroll
  for (int kc = 0; kc < 8; ++kc) {
    const __bf16* p = q_lds + (size_t)(mt + l15) * C_DIM + kc * 32 + base_k;
    V16 t;
    t.u[0] = *reinterpret_cast<const uint4*>(p);        // K run [base_k, +8)
    t.u[1] = *reinterpret_cast<const uint4*>(p + 16);   // K run [base_k+16, +8)
    aq[kc] = t.v;
  }

  float sum8[8] = {0.f,0.f,0.f,0.f,0.f,0.f,0.f,0.f};

  // ---- scores pass: each wave covers 16 rows x (its 32 n-tiles of 16 cols)
  for (int j = 0; j < S_DIM / 64; ++j) {
    const int s0 = ng * 16 + j * 64;
    // B 32x16 layout: lane n=l15, elems e: K = lhalf*8 + e + (e>=8 ? 8 : 0), matching A
    const __bf16* krow = khat + ((size_t)b * S_DIM + s0 + l15) * C_DIM + lhalf * 8;
    __builtin_prefetch(krow + (size_t)64 * C_DIM, 0, 1);
    v8f acc = {0.f,0.f,0.f,0.f,0.f,0.f,0.f,0.f};
    #pragma unroll
    for (int kc = 0; kc < 8; ++kc) {
      V16 bm;
      bm.u[0] = *reinterpret_cast<const uint4*>(krow + kc * 32);
      bm.u[1] = *reinterpret_cast<const uint4*>(krow + kc * 32 + 16);
      acc = __builtin_amdgcn_wmma_f32_16x16x32_bf16(false, aq[kc], false, bm.v,
                                                    (short)0, acc, false, false);
    }
    const float valid = mask[(size_t)b * S_DIM + s0 + l15] ? 0.0f : 1.0f;
    #pragma unroll
    for (int r = 0; r < 8; ++r) {
      float sc = acc[r];
      sc *= sc;                                // fidelity = dot^2, in [0,1]
      const float ev = valid * __expf(sc);     // no max-subtraction needed
      sum8[r] += ev;
      p_lds[(size_t)(mt + r + lhalf * 8) * S_DIM + s0 + l15] = (__bf16)ev;
    }
  }

  // ---- row-sum reduction: across the 16-lane group, then across the 4 waves
  #pragma unroll
  for (int r = 0; r < 8; ++r) {
    float v = sum8[r];
    v += __shfl_xor(v, 1, 32);
    v += __shfl_xor(v, 2, 32);
    v += __shfl_xor(v, 4, 32);
    v += __shfl_xor(v, 8, 32);
    if (l15 == 0) atomicAdd(&rowsum[mt + r + lhalf * 8], v);   // ds_add_f32
  }
  __syncthreads();
  if (tid < M_TILE) {
    const float s = rowsum[tid];
    invz[tid] = (s > 0.0f) ? (1.0f / s) : 0.0f;   // sum==0 => fully masked row
  }
  __syncthreads();

  // ---- Phase B: attention weights (T,B,S) f32 ----
  for (int row = 0; row < M_TILE; ++row) {
    const float iz = invz[row];
    __attribute__((aligned(16))) __bf16 tmp[8];
    *reinterpret_cast<uint4*>(tmp) =
        *reinterpret_cast<const uint4*>(p_lds + (size_t)row * S_DIM + tid * 8);
    float* op = attn + (((size_t)(m0 + row)) * B_DIM + b) * S_DIM + tid * 8;
    float4 o0 = { (float)tmp[0]*iz, (float)tmp[1]*iz, (float)tmp[2]*iz, (float)tmp[3]*iz };
    float4 o1 = { (float)tmp[4]*iz, (float)tmp[5]*iz, (float)tmp[6]*iz, (float)tmp[7]*iz };
    *reinterpret_cast<float4*>(op)     = o0;
    *reinterpret_cast<float4*>(op + 4) = o1;
  }

  // ---- Phase C: context = (P @ V) * invz, B-operand from vT (B,C,S) ----
  const int cg = wave >> 1;
  for (int jc = 0; jc < 4; ++jc) {
    const int c0 = cg * 16 + jc * 64;
    v8f acc = {0.f,0.f,0.f,0.f,0.f,0.f,0.f,0.f};
    const __bf16* brow = vT + ((size_t)b * C_DIM + c0 + l15) * S_DIM + lhalf * 8;
    #pragma unroll 4
    for (int scc = 0; scc < S_DIM / 32; ++scc) {
      const int s0 = scc * 32;
      const __bf16* ap = p_lds + (size_t)(mt + l15) * S_DIM + s0 + base_k;
      V16 am, bm;
      am.u[0] = *reinterpret_cast<const uint4*>(ap);        // ds_load_b128
      am.u[1] = *reinterpret_cast<const uint4*>(ap + 16);
      bm.u[0] = *reinterpret_cast<const uint4*>(brow + s0);
      bm.u[1] = *reinterpret_cast<const uint4*>(brow + s0 + 16);
      acc = __builtin_amdgcn_wmma_f32_16x16x32_bf16(false, am.v, false, bm.v,
                                                    (short)0, acc, false, false);
    }
    #pragma unroll
    for (int r = 0; r < 8; ++r) {
      const int row = mt + r + lhalf * 8;
      ctx[(((size_t)(m0 + row)) * B_DIM + b) * C_DIM + c0 + l15] = acc[r] * invz[row];
    }
  }
}

// ---------------------------------------------------------------------------
extern "C" void kernel_launch(void* const* d_in, const int* in_sizes, int n_in,
                              void* d_out, int out_size, void* d_ws, size_t ws_size,
                              hipStream_t stream) {
  const float* query = (const float*)d_in[0];            // (T,B,C)
  const float* key   = (const float*)d_in[1];            // (S,B,C)
  const float* value = (const float*)d_in[2];            // (S,B,C)
  const unsigned char* mask = (const unsigned char*)d_in[3]; // (B,S) bool

  char* ws = (char*)d_ws;
  const size_t qbytes = (size_t)B_DIM * T_DIM * C_DIM * 2;
  const size_t kbytes = (size_t)B_DIM * S_DIM * C_DIM * 2;
  __bf16* qhat = (__bf16*)ws;                            // (B,T,C) bf16
  __bf16* khat = (__bf16*)(ws + qbytes);                 // (B,S,C) bf16
  __bf16* vT   = (__bf16*)(ws + qbytes + kbytes);        // (B,C,S) bf16

  float* ctx  = (float*)d_out;                           // (T,B,C)
  float* attn = (float*)d_out + (size_t)T_DIM * B_DIM * C_DIM; // (T,B,S)

  qa_norm_kernel<<<(T_DIM * B_DIM) / 8, 256, 0, stream>>>(query, qhat, T_DIM);
  qa_norm_kernel<<<(S_DIM * B_DIM) / 8, 256, 0, stream>>>(key,   khat, S_DIM);
  qa_vt_kernel<<<B_DIM * (S_DIM / 64) * (C_DIM / 64), 256, 0, stream>>>(value, vT);

  const size_t smem = (size_t)M_TILE * C_DIM * 2   // Q tile
                    + (size_t)M_TILE * S_DIM * 2   // probs tile
                    + 2 * M_TILE * sizeof(float);  // rowsum + invz
  qa_attn_kernel<<<B_DIM * (T_DIM / M_TILE), 256, smem, stream>>>(
      qhat, khat, vT, mask, ctx, attn);
}
